// _LSEVanillaAttention_10342281249006
// MI455X (gfx1250) — matrix operand order repaired and checked by reading, not compiled, not run on hardware
//
#include <hip/hip_runtime.h>
#include <math.h>

// ---------------------------------------------------------------------------
// Flash attention forward (with LSE) for MI455X / gfx1250, wave32 + WMMA bf16.
// B=2, S=2048, H=16, D=128, fp32 in/out, bf16 matrix math, fp32 accumulate.
// v2: bf16 prepass for K / V^T into d_ws, hot loop stages tiles with
//     global_load_async_to_lds_b128 + double-buffered LDS (ASYNCcnt pipelined).
// ---------------------------------------------------------------------------

typedef __attribute__((ext_vector_type(16))) __bf16 v16bf;
typedef __attribute__((ext_vector_type(8)))  float  v8f;

constexpr int B_ = 2, S_ = 2048, H_ = 16, D_ = 128;
constexpr int BLOCK_M = 128;   // queries per workgroup (8 waves x 16)
constexpr int BLOCK_N = 32;    // keys per iteration (one 16x16x32 PV K-dim)
constexpr size_t NELEM = (size_t)B_ * S_ * H_ * D_;
// fold 1/sqrt(D) and log2(e) into Q so we can use exp2 in the softmax
constexpr float SCALE_LOG2E = 0.08838834764831845f * 1.4426950408889634f;

struct U32x8 { uint4 lo, hi; };

__device__ __forceinline__ v16bf as_v16bf(uint4 lo, uint4 hi) {
  U32x8 t{lo, hi};
  return __builtin_bit_cast(v16bf, t);
}

__device__ __forceinline__ unsigned bf16_1(float a) {   // f32 -> bf16, rnd-half-up
  return (__float_as_uint(a) + 0x8000u) >> 16;
}
__device__ __forceinline__ unsigned bf16pk(float a, float b) {
  return bf16_1(a) | (bf16_1(b) << 16);
}

// cross-half (lanes 0-15 <-> 16-31) exchange: ds_swizzle SWAPX16
__device__ __forceinline__ float swz16(float x) {
  return __int_as_float(__builtin_amdgcn_ds_swizzle(__float_as_int(x), 0x401F));
}

// ---- CDNA5 async global->LDS copy (16B per lane), tracked by ASYNCcnt ------
__device__ __forceinline__ void async_copy16(unsigned lds_off, const void* gaddr) {
  asm volatile("global_load_async_to_lds_b128 %0, %1, off"
               :: "v"(lds_off), "v"(gaddr) : "memory");
}
__device__ __forceinline__ void wait_async0() {
  asm volatile("s_wait_asynccnt 0x0" ::: "memory");
}
__device__ __forceinline__ void wait_async4() {   // previous tile (4 ops) done
  asm volatile("s_wait_asynccnt 0x4" ::: "memory");
}
__device__ __forceinline__ unsigned lds_off32(const void* p) {
  return (unsigned)(size_t)p;   // low 32 bits of generic addr == LDS byte offset
}

// ============================================================================
// Prepass 1: K (f32, [B,S,H,D]) -> Kbf (bf16, same layout). Pure copy-convert.
// ============================================================================
__global__ __launch_bounds__(256) void k_convert_kernel(
    const float* __restrict__ K, unsigned short* __restrict__ Kbf) {
  const size_t i = ((size_t)blockIdx.x * 256 + threadIdx.x) * 8;
  const float4* p = (const float4*)(K + i);
  const float4 a = p[0], b = p[1];
  *(uint4*)(Kbf + i) = make_uint4(bf16pk(a.x, a.y), bf16pk(a.z, a.w),
                                  bf16pk(b.x, b.y), bf16pk(b.z, b.w));
}

// ============================================================================
// Prepass 2: V (f32, [B,S,H,D]) -> VT (bf16, [B,H,D,S]) via LDS 64x64 tiles.
// ============================================================================
__global__ __launch_bounds__(256) void v_transpose_kernel(
    const float* __restrict__ V, unsigned short* __restrict__ VT) {
  __shared__ __align__(16) unsigned short t[64][72];   // 144B rows, 16B aligned
  const int tid = threadIdx.x;
  int bid = blockIdx.x;
  const int ndt = D_ / 64, nst = S_ / 64;
  const int dt = bid % ndt; bid /= ndt;
  const int st = bid % nst; bid /= nst;
  const int h = bid % H_;
  const int b = bid / H_;
  const int s0 = st * 64, d0 = dt * 64;
  {
    const int r = tid >> 2, seg = tid & 3;
    const float4* src =
        (const float4*)(V + ((size_t)(b * S_ + s0 + r) * H_ + h) * D_ + d0 + seg * 16);
    const float4 a0 = src[0], a1 = src[1], a2 = src[2], a3 = src[3];
    uint4* dst = (uint4*)&t[r][seg * 16];
    dst[0] = make_uint4(bf16pk(a0.x, a0.y), bf16pk(a0.z, a0.w),
                        bf16pk(a1.x, a1.y), bf16pk(a1.z, a1.w));
    dst[1] = make_uint4(bf16pk(a2.x, a2.y), bf16pk(a2.z, a2.w),
                        bf16pk(a3.x, a3.y), bf16pk(a3.z, a3.w));
  }
  __syncthreads();
  {
    const int d = tid >> 2, seg = tid & 3;
    unsigned short* orow =
        VT + ((size_t)(b * H_ + h) * D_ + d0 + d) * S_ + s0 + seg * 16;
    unsigned w[8];
#pragma unroll
    for (int j = 0; j < 8; ++j)
      w[j] = (unsigned)t[seg * 16 + 2 * j][d] |
             ((unsigned)t[seg * 16 + 2 * j + 1][d] << 16);
    ((uint4*)orow)[0] = make_uint4(w[0], w[1], w[2], w[3]);
    ((uint4*)orow)[1] = make_uint4(w[4], w[5], w[6], w[7]);
  }
}

// ============================================================================
// Main kernel: async-staged, double-buffered flash attention.
// ============================================================================
__global__ __launch_bounds__(256) void fattn_main_kernel(
    const float* __restrict__ Q, const unsigned short* __restrict__ Kbf,
    const unsigned short* __restrict__ VT, float* __restrict__ Out,
    float* __restrict__ Lse) {
  __shared__ __align__(16) unsigned short lds_k[2][BLOCK_N * D_];  // [key][d]
  __shared__ __align__(16) unsigned short lds_v[2][D_ * BLOCK_N];  // [d][key]

  const int tid  = threadIdx.x;
  const int lane = tid & 31;
  const int wave = tid >> 5;
  const int hl   = lane >> 4;
  const int ln16 = lane & 15;

  const int qblocks = S_ / BLOCK_M;
  const int qb = blockIdx.x % qblocks;
  const int bh = blockIdx.x / qblocks;
  const int h  = bh % H_;
  const int b  = bh / H_;
  const int q_glob = qb * BLOCK_M + wave * 16 + ln16;

  // ---- B-operand of K*Q^T (lane N = q), pre-scaled by 1/sqrt(D)*log2(e) ----
  v16bf bq[4];
  {
    const float* qrow = Q + ((size_t)(b * S_ + q_glob) * H_ + h) * D_;
#pragma unroll
    for (int c = 0; c < 4; ++c) {
      const float4* p = (const float4*)(qrow + 32 * c + 8 * hl);
      const float4 f0 = p[0], f1 = p[1], f2 = p[4], f3 = p[5];
      bq[c] = as_v16bf(
          make_uint4(bf16pk(f0.x * SCALE_LOG2E, f0.y * SCALE_LOG2E),
                     bf16pk(f0.z * SCALE_LOG2E, f0.w * SCALE_LOG2E),
                     bf16pk(f1.x * SCALE_LOG2E, f1.y * SCALE_LOG2E),
                     bf16pk(f1.z * SCALE_LOG2E, f1.w * SCALE_LOG2E)),
          make_uint4(bf16pk(f2.x * SCALE_LOG2E, f2.y * SCALE_LOG2E),
                     bf16pk(f2.z * SCALE_LOG2E, f2.w * SCALE_LOG2E),
                     bf16pk(f3.x * SCALE_LOG2E, f3.y * SCALE_LOG2E),
                     bf16pk(f3.z * SCALE_LOG2E, f3.w * SCALE_LOG2E)));
    }
  }

  float m_run = -INFINITY, l_run = 0.f;
  v8f acc[8];
#pragma unroll
  for (int i = 0; i < 8; ++i)
#pragma unroll
    for (int r = 0; r < 8; ++r) acc[i][r] = 0.f;

  // ---- per-thread async staging chunks (4 x b128 per thread per tile) ----
  const unsigned short* kplane = Kbf + ((size_t)b * S_ * H_ + h) * D_;
  const unsigned short* vplane = VT + (size_t)(b * H_ + h) * D_ * S_;
  const int kr0 = tid >> 4,          kp0 = tid & 15;          // K rows 0..15
  const int kr1 = (tid + 256) >> 4,  kp1 = tid & 15;          // K rows 16..31
  const int vd0 = tid >> 2,          vp0 = tid & 3;           // V^T d 0..63
  const int vd1 = (tid + 256) >> 2,  vp1 = tid & 3;           // V^T d 64..127

  auto issue_tile = [&](int kb, int buf) {
    async_copy16(lds_off32(&lds_k[buf][kr0 * D_ + kp0 * 8]),
                 kplane + (size_t)(kb + kr0) * (H_ * D_) + kp0 * 8);
    async_copy16(lds_off32(&lds_k[buf][kr1 * D_ + kp1 * 8]),
                 kplane + (size_t)(kb + kr1) * (H_ * D_) + kp1 * 8);
    async_copy16(lds_off32(&lds_v[buf][vd0 * BLOCK_N + vp0 * 8]),
                 vplane + (size_t)vd0 * S_ + kb + vp0 * 8);
    async_copy16(lds_off32(&lds_v[buf][vd1 * BLOCK_N + vp1 * 8]),
                 vplane + (size_t)vd1 * S_ + kb + vp1 * 8);
  };

  issue_tile(0, 0);

  for (int kb = 0, it = 0; kb < S_; kb += BLOCK_N, ++it) {
    const int buf = it & 1;
    const bool more = (kb + BLOCK_N) < S_;
    if (more) issue_tile(kb + BLOCK_N, buf ^ 1);   // prefetch next tile
    if (more) wait_async4(); else wait_async0();   // this tile's 4 ops done
    __syncthreads();                               // all waves' copies landed

    // ============ S^T tiles = K_tile x Q^T (lane N = q, VGPR = key) =========
    v8f st0 = {0.f, 0.f, 0.f, 0.f, 0.f, 0.f, 0.f, 0.f};
    v8f st1 = st0;
#pragma unroll
    for (int c = 0; c < 4; ++c) {      // A layout: runs of 8 at d=32c+8*hl, +16
      const unsigned base = (unsigned)(ln16 * D_ + 32 * c + 8 * hl);
      v16bf ak = as_v16bf(*(const uint4*)&lds_k[buf][base],
                          *(const uint4*)&lds_k[buf][base + 16]);
      st0 = __builtin_amdgcn_wmma_f32_16x16x32_bf16(false, ak, false, bq[c],
                                                    (short)0, st0, false, false);
    }
#pragma unroll
    for (int c = 0; c < 4; ++c) {
      const unsigned base = (unsigned)((16 + ln16) * D_ + 32 * c + 8 * hl);
      v16bf ak = as_v16bf(*(const uint4*)&lds_k[buf][base],
                          *(const uint4*)&lds_k[buf][base + 16]);
      st1 = __builtin_amdgcn_wmma_f32_16x16x32_bf16(false, ak, false, bq[c],
                                                    (short)0, st1, false, false);
    }

    // ======================= online softmax (base-2) =======================
    float bmax = -INFINITY;
#pragma unroll
    for (int r = 0; r < 8; ++r) bmax = fmaxf(bmax, fmaxf(st0[r], st1[r]));
    bmax = fmaxf(bmax, swz16(bmax));
    const float m_new = fmaxf(m_run, bmax);
    const float corr  = exp2f(m_run - m_new);

    float p0[8], p1[8], rsum = 0.f;
#pragma unroll
    for (int r = 0; r < 8; ++r) {
      p0[r] = exp2f(st0[r] - m_new);
      p1[r] = exp2f(st1[r] - m_new);
      rsum += p0[r] + p1[r];
    }
    rsum += swz16(rsum);
    l_run = l_run * corr + rsum;
    m_run = m_new;
#pragma unroll
    for (int i = 0; i < 8; ++i) acc[i] = acc[i] * corr;

    // ==== build P^T B-operand: half0 keys 0..15, half1 keys 16..31 ==========
    unsigned bpw[8];
#pragma unroll
    for (int j = 0; j < 4; ++j) {
      const float sA = swz16(p1[2 * j]);
      const float sB = swz16(p1[2 * j + 1]);
      bpw[j] = bf16pk((hl == 0) ? p0[2 * j] : sA,
                      (hl == 0) ? p0[2 * j + 1] : sB);
    }
#pragma unroll
    for (int j = 0; j < 4; ++j) {
      const float sA = swz16(p0[2 * j]);
      const float sB = swz16(p0[2 * j + 1]);
      bpw[4 + j] = bf16pk((hl == 0) ? sA : p1[2 * j],
                          (hl == 0) ? sB : p1[2 * j + 1]);
    }
    const v16bf bp = as_v16bf(make_uint4(bpw[0], bpw[1], bpw[2], bpw[3]),
                              make_uint4(bpw[4], bpw[5], bpw[6], bpw[7]));

    // ================== O^T += V^T x P^T (8 d-tiles of 16) ==================
#pragma unroll
    for (int dt = 0; dt < 8; ++dt) {
      const unsigned base = (unsigned)((16 * dt + ln16) * BLOCK_N + 16 * hl);
      v16bf av = as_v16bf(*(const uint4*)&lds_v[buf][base],
                          *(const uint4*)&lds_v[buf][base + 8]);
      acc[dt] = __builtin_amdgcn_wmma_f32_16x16x32_bf16(false, av, false, bp,
                                                        (short)0, acc[dt],
                                                        false, false);
    }
    __syncthreads();   // everyone done with buf before it is refilled
  }

  // ============================ epilogue ====================================
  const float inv_l = 1.0f / l_run;
  float* orow = Out + ((size_t)(b * S_ + q_glob) * H_ + h) * D_ + 8 * hl;
#pragma unroll
  for (int dt = 0; dt < 8; ++dt) {
    float4 o0, o1;
    o0.x = acc[dt][0] * inv_l; o0.y = acc[dt][1] * inv_l;
    o0.z = acc[dt][2] * inv_l; o0.w = acc[dt][3] * inv_l;
    o1.x = acc[dt][4] * inv_l; o1.y = acc[dt][5] * inv_l;
    o1.z = acc[dt][6] * inv_l; o1.w = acc[dt][7] * inv_l;
    ((float4*)(orow + 16 * dt))[0] = o0;
    ((float4*)(orow + 16 * dt))[1] = o1;
  }
  if (lane < 16) {
    Lse[((size_t)b * H_ + h) * S_ + q_glob] =
        0.6931471805599453f * (m_run + log2f(l_run));
  }
}

// ============================================================================
// Fallback (round-1 single-pass kernel) in case ws_size is too small.
// ============================================================================
__global__ __launch_bounds__(256) void fattn_fallback_kernel(
    const float* __restrict__ Q, const float* __restrict__ K,
    const float* __restrict__ V, float* __restrict__ Out,
    float* __restrict__ Lse) {
  __shared__ __align__(16) unsigned short lds_k[BLOCK_N * D_];
  __shared__ __align__(16) unsigned short lds_v[D_ * BLOCK_N];

  const int tid  = threadIdx.x;
  const int lane = tid & 31;
  const int wave = tid >> 5;
  const int hl   = lane >> 4;
  const int ln16 = lane & 15;

  const int qblocks = S_ / BLOCK_M;
  const int qb = blockIdx.x % qblocks;
  const int bh = blockIdx.x / qblocks;
  const int h  = bh % H_;
  const int b  = bh / H_;
  const int q_glob = qb * BLOCK_M + wave * 16 + ln16;
  const size_t row_stride = (size_t)H_ * D_;

  v16bf bq[4];
  {
    const float* qrow = Q + ((size_t)(b * S_ + q_glob) * H_ + h) * D_;
#pragma unroll
    for (int c = 0; c < 4; ++c) {
      const float4* p = (const float4*)(qrow + 32 * c + 8 * hl);
      const float4 f0 = p[0], f1 = p[1], f2 = p[4], f3 = p[5];
      bq[c] = as_v16bf(
          make_uint4(bf16pk(f0.x * SCALE_LOG2E, f0.y * SCALE_LOG2E),
                     bf16pk(f0.z * SCALE_LOG2E, f0.w * SCALE_LOG2E),
                     bf16pk(f1.x * SCALE_LOG2E, f1.y * SCALE_LOG2E),
                     bf16pk(f1.z * SCALE_LOG2E, f1.w * SCALE_LOG2E)),
          make_uint4(bf16pk(f2.x * SCALE_LOG2E, f2.y * SCALE_LOG2E),
                     bf16pk(f2.z * SCALE_LOG2E, f2.w * SCALE_LOG2E),
                     bf16pk(f3.x * SCALE_LOG2E, f3.y * SCALE_LOG2E),
                     bf16pk(f3.z * SCALE_LOG2E, f3.w * SCALE_LOG2E)));
    }
  }

  float m_run = -INFINITY, l_run = 0.f;
  v8f acc[8];
#pragma unroll
  for (int i = 0; i < 8; ++i)
#pragma unroll
    for (int r = 0; r < 8; ++r) acc[i][r] = 0.f;

  const float* kbase = K + ((size_t)b * S_ * H_ + h) * D_;
  const float* vbase = V + ((size_t)b * S_ * H_ + h) * D_;

  for (int kb = 0; kb < S_; kb += BLOCK_N) {
    {
      const int row = tid >> 3;
      const int d0  = (tid & 7) * 16;
      const float4* kr = (const float4*)(kbase + (size_t)(kb + row) * row_stride + d0);
      const float4* vr = (const float4*)(vbase + (size_t)(kb + row) * row_stride + d0);
      const float4 a0 = kr[0], a1 = kr[1], a2 = kr[2], a3 = kr[3];
      ((uint4*)&lds_k[row * D_ + d0])[0] =
          make_uint4(bf16pk(a0.x, a0.y), bf16pk(a0.z, a0.w),
                     bf16pk(a1.x, a1.y), bf16pk(a1.z, a1.w));
      ((uint4*)&lds_k[row * D_ + d0])[1] =
          make_uint4(bf16pk(a2.x, a2.y), bf16pk(a2.z, a2.w),
                     bf16pk(a3.x, a3.y), bf16pk(a3.z, a3.w));
      const float4 b0 = vr[0], b1 = vr[1], b2 = vr[2], b3 = vr[3];
      const float vv[16] = {b0.x, b0.y, b0.z, b0.w, b1.x, b1.y, b1.z, b1.w,
                            b2.x, b2.y, b2.z, b2.w, b3.x, b3.y, b3.z, b3.w};
#pragma unroll
      for (int i = 0; i < 16; ++i)
        lds_v[(d0 + i) * BLOCK_N + row] = (unsigned short)bf16_1(vv[i]);
    }
    __syncthreads();

    v8f st0 = {0.f, 0.f, 0.f, 0.f, 0.f, 0.f, 0.f, 0.f};
    v8f st1 = st0;
#pragma unroll
    for (int c = 0; c < 4; ++c) {
      const unsigned base = (unsigned)(ln16 * D_ + 32 * c + 8 * hl);
      v16bf ak = as_v16bf(*(const uint4*)&lds_k[base],
                          *(const uint4*)&lds_k[base + 16]);
      st0 = __builtin_amdgcn_wmma_f32_16x16x32_bf16(false, ak, false, bq[c],
                                                    (short)0, st0, false, false);
    }
#pragma unroll
    for (int c = 0; c < 4; ++c) {
      const unsigned base = (unsigned)((16 + ln16) * D_ + 32 * c + 8 * hl);
      v16bf ak = as_v16bf(*(const uint4*)&lds_k[base],
                          *(const uint4*)&lds_k[base + 16]);
      st1 = __builtin_amdgcn_wmma_f32_16x16x32_bf16(false, ak, false, bq[c],
                                                    (short)0, st1, false, false);
    }

    float bmax = -INFINITY;
#pragma unroll
    for (int r = 0; r < 8; ++r) bmax = fmaxf(bmax, fmaxf(st0[r], st1[r]));
    bmax = fmaxf(bmax, swz16(bmax));
    const float m_new = fmaxf(m_run, bmax);
    const float corr  = exp2f(m_run - m_new);

    float p0[8], p1[8], rsum = 0.f;
#pragma unroll
    for (int r = 0; r < 8; ++r) {
      p0[r] = exp2f(st0[r] - m_new);
      p1[r] = exp2f(st1[r] - m_new);
      rsum += p0[r] + p1[r];
    }
    rsum += swz16(rsum);
    l_run = l_run * corr + rsum;
    m_run = m_new;
#pragma unroll
    for (int i = 0; i < 8; ++i) acc[i] = acc[i] * corr;

    unsigned bpw[8];
#pragma unroll
    for (int j = 0; j < 4; ++j) {
      const float sA = swz16(p1[2 * j]);
      const float sB = swz16(p1[2 * j + 1]);
      bpw[j] = bf16pk((hl == 0) ? p0[2 * j] : sA,
                      (hl == 0) ? p0[2 * j + 1] : sB);
    }
#pragma unroll
    for (int j = 0; j < 4; ++j) {
      const float sA = swz16(p0[2 * j]);
      const float sB = swz16(p0[2 * j + 1]);
      bpw[4 + j] = bf16pk((hl == 0) ? sA : p1[2 * j],
                          (hl == 0) ? sB : p1[2 * j + 1]);
    }
    const v16bf bp = as_v16bf(make_uint4(bpw[0], bpw[1], bpw[2], bpw[3]),
                              make_uint4(bpw[4], bpw[5], bpw[6], bpw[7]));

#pragma unroll
    for (int dt = 0; dt < 8; ++dt) {
      const unsigned base = (unsigned)((16 * dt + ln16) * BLOCK_N + 16 * hl);
      v16bf av = as_v16bf(*(const uint4*)&lds_v[base],
                          *(const uint4*)&lds_v[base + 8]);
      acc[dt] = __builtin_amdgcn_wmma_f32_16x16x32_bf16(false, av, false, bp,
                                                        (short)0, acc[dt],
                                                        false, false);
    }
    __syncthreads();
  }

  const float inv_l = 1.0f / l_run;
  float* orow = Out + ((size_t)(b * S_ + q_glob) * H_ + h) * D_ + 8 * hl;
#pragma unroll
  for (int dt = 0; dt < 8; ++dt) {
    float4 o0, o1;
    o0.x = acc[dt][0] * inv_l; o0.y = acc[dt][1] * inv_l;
    o0.z = acc[dt][2] * inv_l; o0.w = acc[dt][3] * inv_l;
    o1.x = acc[dt][4] * inv_l; o1.y = acc[dt][5] * inv_l;
    o1.z = acc[dt][6] * inv_l; o1.w = acc[dt][7] * inv_l;
    ((float4*)(orow + 16 * dt))[0] = o0;
    ((float4*)(orow + 16 * dt))[1] = o1;
  }
  if (lane < 16) {
    Lse[((size_t)b * H_ + h) * S_ + q_glob] =
        0.6931471805599453f * (m_run + log2f(l_run));
  }
}

extern "C" void kernel_launch(void* const* d_in, const int* /*in_sizes*/,
                              int /*n_in*/, void* d_out, int /*out_size*/,
                              void* d_ws, size_t ws_size, hipStream_t stream) {
  const float* q = (const float*)d_in[0];
  const float* k = (const float*)d_in[1];
  const float* v = (const float*)d_in[2];
  float* out = (float*)d_out;
  float* lse = out + NELEM;
  const dim3 grid(B_ * H_ * (S_ / BLOCK_M));

  const size_t need = 2 * NELEM * sizeof(unsigned short);   // Kbf + VT
  if (ws_size >= need) {
    unsigned short* kbf = (unsigned short*)d_ws;
    unsigned short* vt  = kbf + NELEM;
    k_convert_kernel<<<dim3((unsigned)(NELEM / 8 / 256)), 256, 0, stream>>>(k, kbf);
    v_transpose_kernel<<<dim3(B_ * H_ * (S_ / 64) * (D_ / 64)), 256, 0, stream>>>(v, vt);
    fattn_main_kernel<<<grid, 256, 0, stream>>>(q, kbf, vt, out, lse);
  } else {
    fattn_fallback_kernel<<<grid, 256, 0, stream>>>(q, k, v, out, lse);
  }
}
